// LSSTransform_609885356387
// MI455X (gfx1250) — compile-verified, matrix operand order checked by reading, not run on hardware
//
#include <hip/hip_runtime.h>
#include <hip/hip_bf16.h>
#include <math.h>

typedef __attribute__((ext_vector_type(16))) _Float16 v16h;
typedef __attribute__((ext_vector_type(8)))  _Float16 v8h;
typedef __attribute__((ext_vector_type(8)))  float    v8f;

// ---------------- problem constants ----------------
constexpr int kCIN  = 256;
constexpr int kCOUT = 80;
constexpr int kFH   = 32;
constexpr int kFW   = 88;
constexpr int kD    = 118;           // (60-1)/0.5
constexpr int kNC   = 6;
constexpr int kHW   = kFH * kFW;     // 2816
constexpr int kNPIX = kNC * kHW;     // 16896
constexpr int kNPTS = kNC * kD * kHW;// 1993728
constexpr int kXN = 360, kYN = 360;
constexpr int kOO  = kD + kCOUT;     // 198 depthnet outputs
constexpr int kMO  = 208;            // padded to 13 m-tiles of 16
constexpr int kCP  = 96;             // 80 input channels padded to 96 (3 k-chunks of 32)

__device__ __forceinline__ v16h lss_load_frag(const _Float16* p) {
  // p points at k = g*8 of a 32-half K-chunk.  CDNA5 16-bit A/B layout:
  // halves 0..7  -> k = g*8 + 0..7 ; halves 8..15 -> k = 16 + g*8 + 0..7
  union { v16h f; v8h h[2]; } u;
  u.h[0] = *(const v8h*)(p);
  u.h[1] = *(const v8h*)(p + 16);
  return u.f;
}

// Async 16B global -> LDS copy (CDNA5 GLOBAL_LOAD_ASYNC_TO_LDS_B128, ASYNCcnt).
__device__ __forceinline__ void lss_async_cp16(uint32_t lds_byte_addr, const void* gsrc) {
  asm volatile("global_load_async_to_lds_b128 %0, %1, off"
               :: "v"(lds_byte_addr), "v"(gsrc) : "memory");
}
__device__ __forceinline__ void lss_async_wait() {
  asm volatile("s_wait_asynccnt 0" ::: "memory");
}

// ---------------- weight conversion ----------------
// depthnet weights, chunk-major: wf[(c*kMO + o)*32 + kk] (contiguous 13312B slabs)
__global__ void lss_cvt_dnw(const float* __restrict__ w, _Float16* __restrict__ wf) {
  int idx = blockIdx.x * 256 + threadIdx.x;          // kMO*kCIN
  if (idx >= kMO * kCIN) return;
  int kk = idx & 31;
  int t  = idx >> 5;
  int o  = t % kMO;
  int c  = t / kMO;
  wf[idx] = (_Float16)((o < kOO) ? w[o * kCIN + c * 32 + kk] : 0.0f);
}

__global__ void lss_cvt_convw(const float* __restrict__ w, _Float16* __restrict__ wf) {
  int idx = blockIdx.x * 256 + threadIdx.x;          // 9*80*96
  if (idx >= 9 * kCOUT * kCP) return;
  int ci = idx % kCP;
  int t  = idx / kCP;
  int co = t % kCOUT;
  int k  = t / kCOUT;                                // kh*3+kw
  int kh = k / 3, kw = k % 3;
  float v = (ci < kCOUT) ? w[((co * kCOUT + ci) * 3 + kh) * 3 + kw] : 0.0f;
  wf[idx] = (_Float16)v;
}

// ---------------- 4x4 matrix inverses (Gauss-Jordan) ----------------
__global__ void lss_inv4(const float* __restrict__ aug, const float* __restrict__ l2i,
                         float* __restrict__ out) {
  int t = threadIdx.x;
  if (t >= 2 * kNC) return;
  const float* src = (t < kNC) ? (aug + t * 16) : (l2i + (t - kNC) * 16);
  float a[4][8];
  for (int r = 0; r < 4; ++r)
    for (int c = 0; c < 4; ++c) { a[r][c] = src[r * 4 + c]; a[r][4 + c] = (r == c) ? 1.0f : 0.0f; }
  for (int col = 0; col < 4; ++col) {
    int piv = col; float best = fabsf(a[col][col]);
    for (int r = col + 1; r < 4; ++r)
      if (fabsf(a[r][col]) > best) { best = fabsf(a[r][col]); piv = r; }
    if (piv != col)
      for (int c = 0; c < 8; ++c) { float tmp = a[col][c]; a[col][c] = a[piv][c]; a[piv][c] = tmp; }
    float d = 1.0f / a[col][col];
    for (int c = 0; c < 8; ++c) a[col][c] *= d;
    for (int r = 0; r < 4; ++r) {
      if (r == col) continue;
      float f = a[r][col];
      for (int c = 0; c < 8; ++c) a[r][c] -= f * a[col][c];
    }
  }
  for (int r = 0; r < 4; ++r)
    for (int c = 0; c < 4; ++c) out[t * 16 + r * 4 + c] = a[r][4 + c];
}

// ---------------- geometry -> voxel segment ids ----------------
__global__ void lss_geom(const float* __restrict__ invm, int* __restrict__ seg) {
  int pt = blockIdx.x * 256 + threadIdx.x;
  if (pt >= kNPTS) return;
  int n   = pt / (kD * kHW);
  int rem = pt % (kD * kHW);
  int d   = rem / kHW;
  int r   = rem % kHW;
  int h   = r / kFW, w = r % kFW;

  float x  = (float)w * (703.0f / 87.0f);   // linspace(0, IW-1, FW)
  float y  = (float)h * (255.0f / 31.0f);   // linspace(0, IH-1, FH)
  float dd = 1.0f + 0.5f * (float)d;

  const float* Ai = invm + n * 16;           // inverse img_aug
  const float* Li = invm + (kNC + n) * 16;   // inverse lidar2image
  float q[3];
  for (int i = 0; i < 3; ++i)
    q[i] = Ai[i*4+0]*x + Ai[i*4+1]*y + Ai[i*4+2]*dd + Ai[i*4+3];
  float p0 = q[0]*q[2], p1 = q[1]*q[2], p2 = q[2];
  float g0, g1, g2;
  g0 = Li[0]*p0 + Li[1]*p1 + Li[2]*p2 + Li[3];
  g1 = Li[4]*p0 + Li[5]*p1 + Li[6]*p2 + Li[7];
  g2 = Li[8]*p0 + Li[9]*p1 + Li[10]*p2 + Li[11];

  // gidx = int32 truncation, matching jnp astype(int32)
  int gx = (int)((g0 + 54.0f) / 0.3f);
  int gy = (int)((g1 + 54.0f) / 0.3f);
  int gz = (int)((g2 + 10.0f) / 20.0f);
  bool kept = (gx >= 0) & (gx < kXN) & (gy >= 0) & (gy < kYN) & (gz >= 0) & (gz < 1);
  seg[pt] = kept ? (gx * kYN + gy) : -1;
}

// ---------------- depthnet GEMM: y[198 x 16896] = W x X  (WMMA f16) --------
__global__ __launch_bounds__(128)
void lss_depthnet(const float* __restrict__ img, const _Float16* __restrict__ wf,
                  const float* __restrict__ bias, float* __restrict__ y) {
  __shared__ __align__(16) _Float16 lds_x[64 * kCIN];     // [pix slot][ci]
  __shared__ __align__(16) _Float16 lds_w[kMO * 32];      // [o][k-chunk]
  int tid = threadIdx.x;
  int blk = blockIdx.x;                                   // 264 blocks * 64 px

  for (int i = tid; i < 64 * kCIN; i += 128) {
    int s = i >> 8, ci = i & 255;
    int gp = blk * 64 + s;
    int cam = gp / kHW, rr = gp % kHW;
    lds_x[i] = (_Float16)img[(cam * kCIN + ci) * kHW + rr];
  }

  int lane = tid & 31, wave = tid >> 5;
  int g = lane >> 4, ln = lane & 15;
  uint32_t ldsw = (uint32_t)(size_t)(void*)lds_w;

  v8f acc[13];
  for (int i = 0; i < 13; ++i)
    for (int rr = 0; rr < 8; ++rr) acc[i][rr] = 0.0f;

  for (int c = 0; c < 8; ++c) {
    if (c + 1 < 8) __builtin_prefetch(wf + (c + 1) * kMO * 32, 0, 1);
    __syncthreads();
    // async contiguous slab copy: 13312B = 832 x 16B chunks (ASYNCcnt path)
    {
      const char* wsrc = (const char*)(wf + c * kMO * 32);
      for (int j = tid; j < (kMO * 32 * 2) / 16; j += 128)
        lss_async_cp16(ldsw + (uint32_t)(j * 16), wsrc + j * 16);
      lss_async_wait();
    }
    __syncthreads();
    v16h bfrag = lss_load_frag(&lds_x[(wave * 16 + ln) * kCIN + c * 32 + g * 8]);
    // software-pipelined A fragments: issue load(mt+1) before wmma(mt)
    v16h a_cur = lss_load_frag(&lds_w[ln * 32 + g * 8]);
    for (int mt = 0; mt < 13; ++mt) {
      v16h a_next = a_cur;
      if (mt < 12) a_next = lss_load_frag(&lds_w[((mt + 1) * 16 + ln) * 32 + g * 8]);
      acc[mt] = __builtin_amdgcn_wmma_f32_16x16x32_f16(
          false, a_cur, false, bfrag, (short)0, acc[mt], false, false);
      a_cur = a_next;
    }
  }

  int gp = blk * 64 + wave * 16 + ln;
  for (int mt = 0; mt < 13; ++mt)
    for (int rr = 0; rr < 8; ++rr) {
      int o = mt * 16 + g * 8 + rr;
      if (o < kOO) y[gp * kMO + o] = acc[mt][rr] + bias[o];
    }
}

// ---------------- softmax + lift + scatter (atomic segment-sum) ------------
__global__ __launch_bounds__(128)
void lss_lift_scatter(const float* __restrict__ y, const int* __restrict__ seg,
                      float* __restrict__ bev) {
  __shared__ float yv[kOO];
  __shared__ float dep[kD];
  __shared__ int   sg[kD];
  __shared__ float red[128];
  int tid = threadIdx.x;
  int gp  = blockIdx.x;
  int cam = gp / kHW, rr = gp % kHW;

  for (int i = tid; i < kOO; i += 128) yv[i] = y[gp * kMO + i];
  for (int i = tid; i < kD;  i += 128) sg[i] = seg[(cam * kD + i) * kHW + rr];
  __syncthreads();

  float m = -3.4e38f;
  for (int i = tid; i < kD; i += 128) m = fmaxf(m, yv[i]);
  red[tid] = m; __syncthreads();
  for (int s = 64; s; s >>= 1) { if (tid < s) red[tid] = fmaxf(red[tid], red[tid + s]); __syncthreads(); }
  float mx = red[0]; __syncthreads();

  float ps = 0.0f;
  for (int i = tid; i < kD; i += 128) { float e = expf(yv[i] - mx); dep[i] = e; ps += e; }
  red[tid] = ps; __syncthreads();
  for (int s = 64; s; s >>= 1) { if (tid < s) red[tid] += red[tid + s]; __syncthreads(); }
  float inv = 1.0f / red[0]; __syncthreads();
  for (int i = tid; i < kD; i += 128) dep[i] *= inv;
  __syncthreads();

  for (int idx = tid; idx < kD * kCOUT; idx += 128) {
    int d = idx / kCOUT, co = idx % kCOUT;
    int s = sg[d];
    if (s >= 0) atomicAdd(&bev[co * (kXN * kYN) + s], dep[d] * yv[kD + co]);
  }
}

// ---------------- 3x3 conv (implicit GEMM, WMMA f16) + BN + ReLU -----------
template <int STRIDE>
__global__ __launch_bounds__(STRIDE == 1 ? 128 : 64)
void lss_conv(const float* __restrict__ in, const _Float16* __restrict__ wf,
              const float* __restrict__ gamma, const float* __restrict__ beta,
              float* __restrict__ out, int inH, int inW, int OH, int OW) {
  constexpr int WAVES = (STRIDE == 1) ? 4 : 2;
  constexpr int TH    = WAVES;                  // output rows per block
  constexpr int IRH   = (TH - 1) * STRIDE + 3;  // staged input rows
  constexpr int IRW   = 15 * STRIDE + 3;        // staged input cols
  __shared__ __align__(16) _Float16 lds_w[kCOUT * kCP];       // [co][ci]
  __shared__ __align__(16) _Float16 lds_in[IRH * IRW * kCP];  // [pix][ci]

  int tid = threadIdx.x;
  int ox0 = blockIdx.x * 16;
  int oy0 = blockIdx.y * TH;
  int iy0 = oy0 * STRIDE - 1, ix0 = ox0 * STRIDE - 1;

  for (int i = tid; i < IRH * IRW * kCP; i += WAVES * 32) {
    int ci = i % kCP;
    int p  = i / kCP;
    int iy = p / IRW, ix = p % IRW;
    int gy = iy0 + iy, gx = ix0 + ix;
    float v = 0.0f;
    if (ci < kCOUT && gy >= 0 && gy < inH && gx >= 0 && gx < inW)
      v = in[(ci * inH + gy) * inW + gx];
    lds_in[i] = (_Float16)v;
  }

  int lane = tid & 31, wave = tid >> 5;
  int g = lane >> 4, ln = lane & 15;
  uint32_t ldsw = (uint32_t)(size_t)(void*)lds_w;

  v8f acc[5];
  for (int i = 0; i < 5; ++i)
    for (int rr = 0; rr < 8; ++rr) acc[i][rr] = 0.0f;

  for (int k = 0; k < 9; ++k) {                 // kh*3+kw
    if (k + 1 < 9) __builtin_prefetch(wf + (k + 1) * kCOUT * kCP, 0, 1);
    __syncthreads();
    // async contiguous weight slab copy: 15360B = 960 x 16B chunks
    {
      const char* wsrc = (const char*)(wf + k * kCOUT * kCP);
      for (int j = tid; j < (kCOUT * kCP * 2) / 16; j += WAVES * 32)
        lss_async_cp16(ldsw + (uint32_t)(j * 16), wsrc + j * 16);
      lss_async_wait();
    }
    __syncthreads();
    int kh = k / 3, kw = k % 3;
    int iy = wave * STRIDE + kh;
    int ix = ln * STRIDE + kw;
    const _Float16* brow = &lds_in[(iy * IRW + ix) * kCP];
    for (int c = 0; c < 3; ++c) {
      v16h bfrag = lss_load_frag(brow + c * 32 + g * 8);
      v16h a_cur = lss_load_frag(&lds_w[ln * kCP + c * 32 + g * 8]);
      for (int mt = 0; mt < 5; ++mt) {
        v16h a_next = a_cur;
        if (mt < 4) a_next = lss_load_frag(&lds_w[((mt + 1) * 16 + ln) * kCP + c * 32 + g * 8]);
        acc[mt] = __builtin_amdgcn_wmma_f32_16x16x32_f16(
            false, a_cur, false, bfrag, (short)0, acc[mt], false, false);
        a_cur = a_next;
      }
    }
  }

  int ox = ox0 + ln;
  int oy = oy0 + wave;
  if (ox < OW && oy < OH) {
    float rs = rsqrtf(1.0f + 1e-5f);
    for (int mt = 0; mt < 5; ++mt)
      for (int rr = 0; rr < 8; ++rr) {
        int co = mt * 16 + g * 8 + rr;
        float v = acc[mt][rr] * (gamma[co] * rs) + beta[co];
        out[(co * OH + oy) * OW + ox] = fmaxf(v, 0.0f);
      }
  }
}

// ---------------- host launcher ----------------
extern "C" void kernel_launch(void* const* d_in, const int* in_sizes, int n_in,
                              void* d_out, int out_size, void* d_ws, size_t ws_size,
                              hipStream_t stream) {
  (void)in_sizes; (void)n_in; (void)out_size; (void)ws_size;
  const float* image_fpn   = (const float*)d_in[0];
  const float* img_aug     = (const float*)d_in[1];
  const float* lidar2image = (const float*)d_in[2];
  const float* depthnet_w  = (const float*)d_in[3];
  const float* depthnet_b  = (const float*)d_in[4];
  const float* conv1_w = (const float*)d_in[5];
  const float* bn1_g   = (const float*)d_in[6];
  const float* bn1_b   = (const float*)d_in[7];
  const float* conv2_w = (const float*)d_in[8];
  const float* bn2_g   = (const float*)d_in[9];
  const float* bn2_b   = (const float*)d_in[10];
  const float* conv3_w = (const float*)d_in[11];
  const float* bn3_g   = (const float*)d_in[12];
  const float* bn3_b   = (const float*)d_in[13];
  float* out = (float*)d_out;

  char* ws = (char*)d_ws;
  size_t off = 0;
  auto alloc = [&](size_t bytes) -> void* {
    void* p = ws + off;
    off = (off + bytes + 255) & ~(size_t)255;
    return p;
  };
  _Float16* wf_dn = (_Float16*)alloc((size_t)kMO * kCIN * 2);
  _Float16* wf_c1 = (_Float16*)alloc((size_t)9 * kCOUT * kCP * 2);
  _Float16* wf_c2 = (_Float16*)alloc((size_t)9 * kCOUT * kCP * 2);
  _Float16* wf_c3 = (_Float16*)alloc((size_t)9 * kCOUT * kCP * 2);
  float*    invm  = (float*)alloc((size_t)2 * kNC * 16 * 4);
  int*      seg   = (int*)alloc((size_t)kNPTS * 4);
  float*    ybuf  = (float*)alloc((size_t)kNPIX * kMO * 4);   // reused as h2 later
  float*    bev   = (float*)alloc((size_t)kCOUT * kXN * kYN * 4);
  float*    h1    = (float*)alloc((size_t)kCOUT * kXN * kYN * 4);
  float*    h2    = ybuf;  // 10.4MB <= 14.1MB, y dead after lift_scatter

  hipMemsetAsync(bev, 0, (size_t)kCOUT * kXN * kYN * 4, stream);

  lss_inv4<<<1, 32, 0, stream>>>(img_aug, lidar2image, invm);
  lss_geom<<<(kNPTS + 255) / 256, 256, 0, stream>>>(invm, seg);

  lss_cvt_dnw<<<(kMO * kCIN + 255) / 256, 256, 0, stream>>>(depthnet_w, wf_dn);
  lss_cvt_convw<<<(9 * kCOUT * kCP + 255) / 256, 256, 0, stream>>>(conv1_w, wf_c1);
  lss_cvt_convw<<<(9 * kCOUT * kCP + 255) / 256, 256, 0, stream>>>(conv2_w, wf_c2);
  lss_cvt_convw<<<(9 * kCOUT * kCP + 255) / 256, 256, 0, stream>>>(conv3_w, wf_c3);

  lss_depthnet<<<kNPIX / 64, 128, 0, stream>>>(image_fpn, wf_dn, depthnet_b, ybuf);
  lss_lift_scatter<<<kNPIX, 128, 0, stream>>>(ybuf, seg, bev);

  lss_conv<1><<<dim3((kYN + 15) / 16, kXN / 4), 128, 0, stream>>>(
      bev, wf_c1, bn1_g, bn1_b, h1, kXN, kYN, kXN, kYN);
  lss_conv<2><<<dim3((180 + 15) / 16, 180 / 2), 64, 0, stream>>>(
      h1, wf_c2, bn2_g, bn2_b, h2, kXN, kYN, 180, 180);
  lss_conv<1><<<dim3((180 + 15) / 16, 180 / 4), 128, 0, stream>>>(
      h2, wf_c3, bn3_g, bn3_b, out, 180, 180, 180, 180);
}
